// SpatialLinearAttention_15865609191616
// MI455X (gfx1250) — matrix operand order repaired and checked by reading, not compiled, not run on hardware
//
#include <hip/hip_runtime.h>
#include <hip/hip_bf16.h>

// ---------------------------------------------------------------------------
// SpatialLinearAttention for MI455X (gfx1250, wave32, WMMA bf16 16x16x32)
// b=2 c=128 f=16 h=w=64  ->  BF=32 images of (C=128, N=4096)
// Memory-bound (~19.4 GFLOP vs ~0.9 GB traffic @ 23.3 TB/s): bf16
// intermediates + v_wmma_f32_16x16x32_bf16 with f32 accumulation.
// x-tile staging uses the Tensor Data Mover (tensor_load_to_lds + TENSORcnt).
// ---------------------------------------------------------------------------

typedef __attribute__((ext_vector_type(16))) __bf16 v16bf;
typedef __attribute__((ext_vector_type(8)))  float  v8f;
typedef __attribute__((ext_vector_type(4)))  unsigned int v4u;
typedef __attribute__((ext_vector_type(4)))  int    v4i;
typedef __attribute__((ext_vector_type(8)))  int    v8i;

union FragAB { v16bf v; unsigned u[8]; unsigned short s[16]; };

__device__ __forceinline__ unsigned short f2bf(float f) {
  unsigned u = __float_as_uint(f);
  u += 0x7FFFu + ((u >> 16) & 1u);               // round-to-nearest-even
  return (unsigned short)(u >> 16);
}
__device__ __forceinline__ float bf2f(unsigned short s) {
  return __uint_as_float(((unsigned)s) << 16);
}
__device__ __forceinline__ unsigned pack_bf2(float lo, float hi) {
  return (unsigned)f2bf(lo) | ((unsigned)f2bf(hi) << 16);
}

// A-fragment (16x32 bf16): lane r=lane&15 is row M; VGPR v holds K pair at:
//   16*half + 2*v
__device__ __forceinline__ int a_base(int v, int half) {
  return 16 * half + 2 * v;
}
// B-fragment (32x16 bf16): lane r=lane&15 is col N; VGPR j holds K pair at
//   16*half + 2*j   (requires K contiguous in memory -> stage transposed)

// Generic LDS pointer low 32 bits == wave-relative LDS byte offset.
__device__ __forceinline__ unsigned lds_off(const void* p) {
  return (unsigned)(unsigned long long)p;
}

// ---------------------------------------------------------------------------
// TDM: 2D tile load Global -> LDS (Tensor DMA descriptor per ISA ch.8).
//   tile_w elements x tile_h rows, rows separated by row_stride elements.
//   ds_log2: element size log2 (0=1B,1=2B,2=4B,3=8B)
// ---------------------------------------------------------------------------
__device__ __forceinline__ void tdm_load_tile_2d(
    unsigned lds_addr, const void* gaddr,
    unsigned tile_w, unsigned tile_h,
    unsigned tensor_w, unsigned tensor_h,
    unsigned long long row_stride, unsigned ds_log2) {
  unsigned long long ga = (unsigned long long)gaddr;
  v4u g0;
  g0.x = 1u;                                            // count=1 (valid D#)
  g0.y = lds_addr;                                      // lds_addr
  g0.z = (unsigned)(ga & 0xFFFFFFFFu);                  // global_addr[31:0]
  g0.w = (unsigned)((ga >> 32) & 0x01FFFFFFu) | (2u << 30);  // ga[56:32]|type=2
  v8i g1;
  g1[0] = (int)(ds_log2 << 16);                         // wg_mask=0, data_size
  g1[1] = (int)((tensor_w & 0xFFFFu) << 16);            // tensor_dim0[15:0]
  g1[2] = (int)(((tensor_w >> 16) & 0xFFFFu) |          // tensor_dim0[31:16]
                ((tensor_h & 0xFFFFu) << 16));          // tensor_dim1[15:0]
  g1[3] = (int)(((tensor_h >> 16) & 0xFFFFu) |          // tensor_dim1[31:16]
                ((tile_w & 0xFFFFu) << 16));            // tile_dim0
  g1[4] = (int)(tile_h & 0xFFFFu);                      // tile_dim1 (dim2=0)
  g1[5] = (int)(row_stride & 0xFFFFFFFFu);              // dim0_stride[31:0]
  g1[6] = (int)((row_stride >> 32) & 0xFFFFu);          // dim0_stride[47:32]
  g1[7] = 0;                                            // dim1_stride unused
  v4i g2 = {0, 0, 0, 0};
  v4i g3 = {0, 0, 0, 0};
#if defined(__clang_major__) && (__clang_major__ >= 23)
  v8i g4 = {0, 0, 0, 0, 0, 0, 0, 0};
  __builtin_amdgcn_tensor_load_to_lds(g0, g1, g2, g3, g4, 0);
#else
  __builtin_amdgcn_tensor_load_to_lds(g0, g1, g2, g3, 0);
#endif
}

// ---------------------------------------------------------------------------
// Kernel 0: convert weights to bf16 (tiny, stays L2-resident)
// ---------------------------------------------------------------------------
__global__ __launch_bounds__(256)
void cvt_weights(const float* __restrict__ Wqkv, const float* __restrict__ Wout,
                 unsigned short* __restrict__ wq, unsigned short* __restrict__ wo) {
  int i = blockIdx.x * 256 + threadIdx.x;
  if (i < 384 * 128) wq[i] = f2bf(Wqkv[i]);
  if (i < 128 * 128) wo[i] = f2bf(Wout[i]);
}

// ---------------------------------------------------------------------------
// Kernel 1: qkv = Wqkv(384x128) @ x(128x32-tile), fused q-softmax epilogue.
// grid(128 n-tiles, 32 images), 256 threads = 8 wave32s.
// x tile staged by the Tensor Data Mover, then converted f32->bf16 transposed.
// ---------------------------------------------------------------------------
__global__ __launch_bounds__(256)
void qkv_kernel(const float* __restrict__ x, const unsigned short* __restrict__ Wq,
                unsigned short* __restrict__ qw, unsigned short* __restrict__ kw,
                unsigned short* __restrict__ vw) {
  __shared__ unsigned short xT[32 * 128];  // [col][c]  (B operand: K contiguous)
  __shared__ float qkv_s[384 * 32];        // staged C; x_raw aliased over start
  float* x_raw = qkv_s;                    // 128x32 f32 TDM landing zone (16KB)
  const int bf = blockIdx.y;
  const int n0 = blockIdx.x * 32;
  const int b  = bf >> 4, f = bf & 15;
  const int tid = threadIdx.x;

  // warm the bf16 weights into cache while the TDM runs
  for (int i = tid; i < 768; i += 256)
    __builtin_prefetch((const char*)Wq + i * 128, 0, 3);

  // TDM: load x tile (128 rows x 32 f32), row stride = 16*4096 elements
  if (tid == 0) {
    const float* gsrc = x + (((size_t)(b * 128) * 16 + f) << 12) + n0;
    tdm_load_tile_2d(lds_off(x_raw), gsrc,
                     /*tile_w=*/32, /*tile_h=*/128,
                     /*tensor_w=*/4096, /*tensor_h=*/128,
                     /*row_stride=*/65536ull, /*ds_log2=*/2u);
    __builtin_amdgcn_s_wait_tensorcnt(0);
  }
  __syncthreads();

  // convert f32 -> bf16, transposed into B-operand layout
  for (int i = 0; i < 16; ++i) {
    int idx = tid + i * 256;
    int c = idx >> 5, col = idx & 31;
    xT[col * 128 + c] = f2bf(x_raw[idx]);
  }
  __syncthreads();   // x_raw dead; qkv_s (aliased) may now be written

  const int wave = tid >> 5, lane = tid & 31;
  const int r = lane & 15, half = lane >> 4;

  for (int t = wave; t < 48; t += 8) {       // 24 row-tiles x 2 col-tiles
    const int tm = t % 24, tn = t / 24;
    v8f acc = {};
    for (int ks = 0; ks < 4; ++ks) {         // K = 128 = 4 x 32
      FragAB a, bx;
      const unsigned short* ap = Wq + (tm * 16 + r) * 128 + ks * 32;
#pragma unroll
      for (int v = 0; v < 8; ++v)
        a.u[v] = *(const unsigned*)(ap + a_base(v, half));
      const unsigned short* bp = xT + (tn * 16 + r) * 128 + ks * 32;
#pragma unroll
      for (int j = 0; j < 8; ++j)
        bx.u[j] = *(const unsigned*)(bp + 16 * half + 2 * j);
      acc = __builtin_amdgcn_wmma_f32_16x16x32_bf16(
          false, a.v, false, bx.v, (short)0, acc, false, false);
    }
#pragma unroll
    for (int j = 0; j < 8; ++j)
      qkv_s[(tm * 16 + j + 8 * half) * 32 + tn * 16 + r] = acc[j];
  }
  __syncthreads();

  // q epilogue: softmax over the 32-channel dim_head axis, * DIM_HEAD^-0.5
  if (tid < 128) {
    const int hd = tid >> 5, col = tid & 31;
    float mx = -1e30f;
    for (int d = 0; d < 32; ++d) mx = fmaxf(mx, qkv_s[(hd * 32 + d) * 32 + col]);
    float sum = 0.f;
    for (int d = 0; d < 32; ++d) sum += __expf(qkv_s[(hd * 32 + d) * 32 + col] - mx);
    const float rs = 0.1767766953f / sum;    // SCALE = 32^-0.5
    for (int d = 0; d < 32; ++d) {
      int ch = hd * 32 + d;
      qw[((size_t)(bf * 128 + ch) << 12) + n0 + col] =
          f2bf(__expf(qkv_s[ch * 32 + col] - mx) * rs);
    }
  }

  // k raw (softmax applied later with global stats) and v / N
  for (int i = 0; i < 32; ++i) {
    int idx = tid + i * 256;                 // 0..8191
    int row = idx >> 5, col = idx & 31;
    if (row < 128) {
      kw[((size_t)(bf * 128 + row) << 12) + n0 + col] =
          f2bf(qkv_s[(128 + row) * 32 + col]);
    } else {
      int vr = row - 128;
      vw[((size_t)(bf * 128 + vr) << 12) + n0 + col] =
          f2bf(qkv_s[(256 + vr) * 32 + col] * (1.0f / 4096.0f));
    }
  }
}

// ---------------------------------------------------------------------------
// Kernel 2: per-(image,channel) rowmax and 1/sum(exp) for k-softmax over N.
// grid(128 channels, 32 images)
// ---------------------------------------------------------------------------
__global__ __launch_bounds__(256)
void kstats_kernel(const unsigned short* __restrict__ kw,
                   float* __restrict__ kmax, float* __restrict__ krsum) {
  __shared__ float red[256];
  const int row = blockIdx.y * 128 + blockIdx.x;
  const int tid = threadIdx.x;
  const unsigned short* p = kw + ((size_t)row << 12);
  float mx = -1e30f;
  for (int i = tid; i < 4096; i += 256) mx = fmaxf(mx, bf2f(p[i]));
  red[tid] = mx; __syncthreads();
  for (int s = 128; s > 0; s >>= 1) {
    if (tid < s) red[tid] = fmaxf(red[tid], red[tid + s]);
    __syncthreads();
  }
  mx = red[0]; __syncthreads();
  float sum = 0.f;
  for (int i = tid; i < 4096; i += 256) sum += __expf(bf2f(p[i]) - mx);
  red[tid] = sum; __syncthreads();
  for (int s = 128; s > 0; s >>= 1) {
    if (tid < s) red[tid] += red[tid + s];
    __syncthreads();
  }
  if (tid == 0) { kmax[row] = mx; krsum[row] = 1.0f / red[0]; }
}

// ---------------------------------------------------------------------------
// Kernel 3: context[d][e] = sum_n ksoft[d,n] * v[e,n]; stored transposed bf16.
// grid(4 heads, 32 images); K=4096 split 8 ways across waves; k-softmax fused.
// ---------------------------------------------------------------------------
__global__ __launch_bounds__(256)
void context_kernel(const unsigned short* __restrict__ kw,
                    const unsigned short* __restrict__ vw,
                    const float* __restrict__ kmax,
                    const float* __restrict__ krsum,
                    unsigned short* __restrict__ ctT) {
  __shared__ float part[8 * 32 * 32];
  const int bf = blockIdx.y, hd = blockIdx.x;
  const int tid = threadIdx.x, wave = tid >> 5, lane = tid & 31;
  const int r = lane & 15, half = lane >> 4;

  v8f zero = {};
  v8f acc[2][2] = {{zero, zero}, {zero, zero}};

  float mx[2], rs[2];
  for (int tm = 0; tm < 2; ++tm) {
    int c = bf * 128 + hd * 32 + tm * 16 + r;
    mx[tm] = kmax[c];
    rs[tm] = krsum[c];
  }

  for (int ks = wave * 16; ks < wave * 16 + 16; ++ks) {  // 128 ksteps / 8 waves
    const int n0k = ks * 32;
    FragAB af[2], bv[2];
    for (int tm = 0; tm < 2; ++tm) {
      const unsigned short* kp =
          kw + ((size_t)(bf * 128 + hd * 32 + tm * 16 + r) << 12) + n0k;
#pragma unroll
      for (int v = 0; v < 8; ++v) {
        unsigned pr = *(const unsigned*)(kp + a_base(v, half));
        float lo = __expf(bf2f((unsigned short)(pr & 0xFFFFu)) - mx[tm]) * rs[tm];
        float hi = __expf(bf2f((unsigned short)(pr >> 16)) - mx[tm]) * rs[tm];
        af[tm].u[v] = pack_bf2(lo, hi);
      }
    }
    for (int tn = 0; tn < 2; ++tn) {
      const unsigned short* vp =
          vw + ((size_t)(bf * 128 + hd * 32 + tn * 16 + r) << 12) + n0k;
#pragma unroll
      for (int j = 0; j < 8; ++j)
        bv[tn].u[j] = *(const unsigned*)(vp + 16 * half + 2 * j);
    }
    for (int tm = 0; tm < 2; ++tm)
      for (int tn = 0; tn < 2; ++tn)
        acc[tm][tn] = __builtin_amdgcn_wmma_f32_16x16x32_bf16(
            false, af[tm].v, false, bv[tn].v, (short)0, acc[tm][tn], false, false);
  }

  for (int tm = 0; tm < 2; ++tm)
    for (int tn = 0; tn < 2; ++tn)
#pragma unroll
      for (int j = 0; j < 8; ++j)
        part[wave * 1024 + (tm * 16 + j + 8 * half) * 32 + tn * 16 + r] =
            acc[tm][tn][j];
  __syncthreads();

  for (int i = tid; i < 1024; i += 256) {
    int d = i >> 5, e = i & 31;
    float s = 0.f;
    for (int w = 0; w < 8; ++w) s += part[w * 1024 + d * 32 + e];
    ctT[(size_t)(bf * 4 + hd) * 1024 + e * 32 + d] = f2bf(s);  // [e][d]
  }
}

// ---------------------------------------------------------------------------
// Kernel 4: out = Wout @ (ctxT @ q) + bout, scattered store in (b,c,f,h,w).
// grid(128 n-tiles, 32 images)
// ---------------------------------------------------------------------------
__global__ __launch_bounds__(256)
void out_kernel(const unsigned short* __restrict__ qw,
                const unsigned short* __restrict__ ctT,
                const unsigned short* __restrict__ WoutBf,
                const float* __restrict__ bout,
                float* __restrict__ out) {
  __shared__ unsigned short q_s[32 * 128];    // [col][d]   (B operand)
  __shared__ unsigned short tmp_s[32 * 128];  // [col][c]   (B operand)
  const int bf = blockIdx.y;
  const int n0 = blockIdx.x * 32;
  const int b  = bf >> 4, f = bf & 15;
  const int tid = threadIdx.x, wave = tid >> 5, lane = tid & 31;
  const int r = lane & 15, half = lane >> 4;

  for (int i = 0; i < 16; ++i) {              // stage q tile transposed
    int idx = tid + i * 256;
    int c = idx >> 5, col = idx & 31;
    q_s[col * 128 + c] = qw[((size_t)(bf * 128 + c) << 12) + n0 + col];
  }
  __syncthreads();

  // step 1: tmp[c][col] = sum_d ctT[e][d] * q[d][col]  (per head, K=32)
  for (int t = wave; t < 16; t += 8) {
    const int tm = t & 7, tn = t >> 3, hd = tm >> 1;
    FragAB a, bq;
    const unsigned short* ap =
        ctT + (size_t)(bf * 4 + hd) * 1024 + ((tm & 1) * 16 + r) * 32;
#pragma unroll
    for (int v = 0; v < 8; ++v)
      a.u[v] = *(const unsigned*)(ap + a_base(v, half));
    const unsigned short* bp = q_s + (tn * 16 + r) * 128 + hd * 32;
#pragma unroll
    for (int j = 0; j < 8; ++j)
      bq.u[j] = *(const unsigned*)(bp + 16 * half + 2 * j);
    v8f acc = {};
    acc = __builtin_amdgcn_wmma_f32_16x16x32_bf16(
        false, a.v, false, bq.v, (short)0, acc, false, false);
#pragma unroll
    for (int j = 0; j < 8; ++j)
      tmp_s[(tn * 16 + r) * 128 + tm * 16 + j + 8 * half] = f2bf(acc[j]);
  }
  __syncthreads();

  // step 2: out = Wout(128x128) @ tmp + bout, K = 128
  for (int t = wave; t < 16; t += 8) {
    const int tm = t & 7, tn = t >> 3;
    v8f acc = {};
    for (int ks = 0; ks < 4; ++ks) {
      FragAB a, bq;
      const unsigned short* ap = WoutBf + (tm * 16 + r) * 128 + ks * 32;
#pragma unroll
      for (int v = 0; v < 8; ++v)
        a.u[v] = *(const unsigned*)(ap + a_base(v, half));
      const unsigned short* bp = tmp_s + (tn * 16 + r) * 128 + ks * 32;
#pragma unroll
      for (int j = 0; j < 8; ++j)
        bq.u[j] = *(const unsigned*)(bp + 16 * half + 2 * j);
      acc = __builtin_amdgcn_wmma_f32_16x16x32_bf16(
          false, a.v, false, bq.v, (short)0, acc, false, false);
    }
#pragma unroll
    for (int j = 0; j < 8; ++j) {
      int c = tm * 16 + j + 8 * half;
      int n = n0 + tn * 16 + r;
      out[(((size_t)(b * 128 + c) * 16 + f) << 12) + n] = acc[j] + bout[c];
    }
  }
}

// ---------------------------------------------------------------------------
extern "C" void kernel_launch(void* const* d_in, const int* in_sizes, int n_in,
                              void* d_out, int out_size, void* d_ws, size_t ws_size,
                              hipStream_t stream) {
  const float* x    = (const float*)d_in[0];
  const float* Wqkv = (const float*)d_in[1];
  const float* Wout = (const float*)d_in[2];
  const float* bout = (const float*)d_in[3];

  char* ws = (char*)d_ws;
  size_t off = 0;
  unsigned short* wq = (unsigned short*)(ws + off); off += 384 * 128 * 2;
  unsigned short* wo = (unsigned short*)(ws + off); off += 128 * 128 * 2;
  off = (off + 255) & ~(size_t)255;
  unsigned short* qw = (unsigned short*)(ws + off); off += (size_t)32 * 128 * 4096 * 2;
  unsigned short* kw = (unsigned short*)(ws + off); off += (size_t)32 * 128 * 4096 * 2;
  unsigned short* vw = (unsigned short*)(ws + off); off += (size_t)32 * 128 * 4096 * 2;
  float* kmax = (float*)(ws + off); off += 32 * 128 * 4;
  float* krs  = (float*)(ws + off); off += 32 * 128 * 4;
  unsigned short* ctT = (unsigned short*)(ws + off); off += (size_t)128 * 1024 * 2;
  // total ~101 MB of workspace

  cvt_weights<<<dim3(192), 256, 0, stream>>>(Wqkv, Wout, wq, wo);
  qkv_kernel<<<dim3(128, 32), 256, 0, stream>>>(x, wq, qw, kw, vw);
  kstats_kernel<<<dim3(128, 32), 256, 0, stream>>>(kw, kmax, krs);
  context_kernel<<<dim3(4, 32), 256, 0, stream>>>(kw, vw, kmax, krs, ctT);
  out_kernel<<<dim3(128, 32), 256, 0, stream>>>(qw, ctT, wo, bout, (float*)d_out);
}
